// sLSTMCell_63754494542061
// MI455X (gfx1250) — matrix operand order repaired and checked by reading, not compiled, not run on hardware
//
#include <hip/hip_runtime.h>
#include <hip/hip_bf16.h>

typedef __bf16 bf16;
typedef bf16  v16bf __attribute__((ext_vector_type(16)));
typedef bf16  v8bf  __attribute__((ext_vector_type(8)));
typedef float v8f   __attribute__((ext_vector_type(8)));

#define L_   1024
#define B_   64
#define DW_  300
#define DH_  256
#define ROWS (L_ * B_)          // 65536
#define COLS 16384              // B_*DH_
#define KTOT 1888               // 5*256 + 320 + 256 + 32
#define NTOT 1792               // 7*256
#define HC_  ((size_t)1025 * COLS)   // offset of c_t in d_out

__device__ __forceinline__ float sigf(float x) {
    return 1.0f / (1.0f + __expf(-x));
}
__device__ __forceinline__ float tanh_f(float x) {
    float ax = fabsf(x);
    float t  = __expf(-2.0f * ax);
    float r  = (1.0f - t) / (1.0f + t);
    return copysignf(r, x);
}

__device__ __forceinline__ v16bf load_frag(const bf16* p0, const bf16* p1) {
    union { v16bf v; v8bf h[2]; } u;
    u.h[0] = *(const v8bf*)p0;
    u.h[1] = *(const v8bf*)p1;
    return u.v;
}

// ---------------------------------------------------------------------------
// prep_h: mask h0/c0 over valid lengths, build bf16 hpad (L+4 rows, zero
// borders), f32 cpad (L+2 rows, zero borders), h_hat mean, hg bf16.
// One thread per (b,d) column; coalesced across l.
// ---------------------------------------------------------------------------
__global__ __launch_bounds__(256) void k_prep_h(
    const float* __restrict__ h0, const float* __restrict__ c0,
    const int* __restrict__ slen,
    bf16* __restrict__ hpad, float* __restrict__ cpad,
    float* __restrict__ hhat, bf16* __restrict__ hgb)
{
    const int tid = blockIdx.x * 256 + threadIdx.x;   // 0..16383  (b*256+d)
    const int b   = tid >> 8;
    const int len = slen[b];
    // zero borders
    hpad[tid] = (bf16)0.0f;
    hpad[(size_t)1 * COLS + tid]    = (bf16)0.0f;
    hpad[(size_t)1026 * COLS + tid] = (bf16)0.0f;
    hpad[(size_t)1027 * COLS + tid] = (bf16)0.0f;
    cpad[tid] = 0.0f;
    cpad[(size_t)1025 * COLS + tid] = 0.0f;

    float sum = 0.0f;
    for (int l = 0; l < L_; ++l) {
        const size_t idx = (size_t)l * COLS + tid;
        const bool ok = (l < len);
        const float hv = ok ? h0[idx] : 0.0f;
        const float cv = ok ? c0[idx] : 0.0f;
        sum += hv;
        hpad[(size_t)(l + 2) * COLS + tid] = (bf16)hv;
        cpad[(size_t)(l + 1) * COLS + tid] = cv;
    }
    hhat[tid] = sum * (1.0f / (float)L_);
    hgb[tid]  = (bf16)h0[(size_t)L_ * COLS + tid];
}

// ---------------------------------------------------------------------------
// prep_src: src (L*B,300) f32 -> srcb (L*B,320) bf16, zero padded.
// ---------------------------------------------------------------------------
__global__ __launch_bounds__(256) void k_prep_src(
    const float* __restrict__ src, bf16* __restrict__ srcb)
{
    const unsigned tid = blockIdx.x * 256u + threadIdx.x;  // < 65536*320
    const unsigned r = tid / 320u;
    const unsigned k = tid % 320u;
    srcb[tid] = (k < DW_) ? (bf16)src[(size_t)r * DW_ + k] : (bf16)0.0f;
}

// ---------------------------------------------------------------------------
// pack_w: build Wcat bf16 (1792 x 1888): [Ww | Wu pad | Wv | Wb onehot pad],
// and Su1b bf16 (256 x 256) = Su[1].
// ---------------------------------------------------------------------------
__global__ __launch_bounds__(256) void k_pack_w(
    const float* __restrict__ Ww, const float* __restrict__ Wu,
    const float* __restrict__ Wv, const float* __restrict__ Wb,
    const float* __restrict__ Su,
    bf16* __restrict__ wcat, bf16* __restrict__ su1b)
{
    const unsigned tid = blockIdx.x * 256u + threadIdx.x;
    const unsigned NW = (unsigned)NTOT * KTOT;   // 3,383,296
    if (tid < NW) {
        const unsigned n = tid / KTOT;   // g*256 + d
        const unsigned k = tid % KTOT;
        float val = 0.0f;
        if (k < 1280u)                      val = Ww[(size_t)n * 1280 + k];
        else if (k < 1580u)                 val = Wu[(size_t)n * 300 + (k - 1280u)];
        else if (k >= 1600u && k < 1856u)   val = Wv[(size_t)n * 256 + (k - 1600u)];
        else if (k == 1856u)                val = Wb[n];
        wcat[tid] = (bf16)val;
    } else if (tid < NW + 65536u) {
        const unsigned s = tid - NW;                 // d*256 + h
        su1b[s] = (bf16)Su[65536u + s];              // Su[1][d][h]
    }
}

// ---------------------------------------------------------------------------
// gates_small: fg = sig(hg@Sw0.T + hhat@Su0.T + Sb0)
//              base = hg@Sw1.T + Sb1   (per-(b,d) part of fi)
//              og = sig(hg@Sw2.T + hhat@Su2.T + Sb2)
// ---------------------------------------------------------------------------
__global__ __launch_bounds__(256) void k_gates_small(
    const float* __restrict__ h0, const float* __restrict__ hhat,
    const float* __restrict__ Sw, const float* __restrict__ Su,
    const float* __restrict__ Sb,
    float* __restrict__ fg, float* __restrict__ og, float* __restrict__ base)
{
    const int tid = blockIdx.x * 256 + threadIdx.x;  // < 3*16384
    const int g   = tid >> 14;
    const int rem = tid & 16383;        // b*256 + d
    const int b   = rem >> 8;
    const int d   = rem & 255;
    const float* hg  = h0 + (size_t)L_ * COLS + (size_t)b * DH_;
    const float* swr = Sw + ((size_t)g * DH_ + d) * DH_;
    const float* sur = Su + ((size_t)g * DH_ + d) * DH_;
    const float* hh  = hhat + (size_t)b * DH_;
    float acc = Sb[g * DH_ + d];
    #pragma unroll 4
    for (int h = 0; h < DH_; ++h) {
        acc += hg[h] * swr[h];
        if (g != 1) acc += hh[h] * sur[h];
    }
    if (g == 0)      fg[rem]   = sigf(acc);
    else if (g == 1) base[rem] = acc;
    else             og[rem]   = sigf(acc);
}

// ---------------------------------------------------------------------------
// fi_gemm (WMMA bf16): fi[l,b,d] (pre-sigmoid, without base) =
//   sum_h h_w1[l,b,h] * Su[1][d,h].  M=65536, K=256, N=256.
// Block = 256 thr = 8 waves, each wave a 16x16 tile; grid (512, 16).
// ---------------------------------------------------------------------------
__global__ __launch_bounds__(256) void k_fi_gemm(
    const bf16* __restrict__ hpad, const bf16* __restrict__ su1b,
    float* __restrict__ fi)
{
    const int tx   = threadIdx.x;
    const int wave = tx >> 5, lane = tx & 31;
    const int half = lane >> 4, lm = lane & 15;
    const int rbase = blockIdx.x * 128 + wave * 16;
    const int d0    = blockIdx.y * 16;
    const int r     = rbase + lm;

    const bf16* arow = hpad + (size_t)(r + 2 * B_) * DH_;  // shift j=2 => h_w1[l]
    const bf16* brow = su1b + (size_t)(d0 + lm) * DH_;

    v8f acc = {};
    #pragma unroll
    for (int kb = 0; kb < 8; ++kb) {
        const int k0 = kb * 32;
        v16bf a = load_frag(arow + k0 + half * 8, arow + k0 + 16 + half * 8);
        v16bf bm = load_frag(brow + k0 + 8 * half, brow + k0 + 16 + 8 * half);
        acc = __builtin_amdgcn_wmma_f32_16x16x32_bf16(
            false, a, false, bm, (short)0, acc, false, false);
    }
    #pragma unroll
    for (int v = 0; v < 8; ++v) {
        const int rr = rbase + v + 8 * half;
        fi[(size_t)rr * DH_ + d0 + lm] = acc[v];
    }
}

// ---------------------------------------------------------------------------
// main_gemm (WMMA bf16): pre[g][l,b,d] for g=0..6 with fused epilogue:
// gate activations, 5-way softmax, c-context combine, mask, write h_w/c_w.
// A row r=(l*64+b), K blocks: j-shifted hpad (5x256) | srcb 320 | hgb 256 |
// one-hot 32 (carries Wb).  Each wave: 16 rows x 16 d x 7 gates.
// ---------------------------------------------------------------------------
__global__ __launch_bounds__(256) void k_main_gemm(
    const bf16* __restrict__ hpad, const bf16* __restrict__ srcb,
    const bf16* __restrict__ hgb,  const bf16* __restrict__ wcat,
    const float* __restrict__ cpad, const float* __restrict__ c0,
    const int* __restrict__ slen, float* __restrict__ out)
{
    const int tx   = threadIdx.x;
    const int wave = tx >> 5, lane = tx & 31;
    const int half = lane >> 4, lm = lane & 15;
    const int rbase = blockIdx.x * 128 + wave * 16;
    const int d0    = blockIdx.y * 16;
    const int r     = rbase + lm;      // A row handled by this lane

    v8f acc[7];
    v8f zero = {};
    #pragma unroll
    for (int g = 0; g < 7; ++g) acc[g] = zero;

    const bf16* brow[7];
    #pragma unroll
    for (int g = 0; g < 7; ++g)
        brow[g] = wcat + (size_t)(g * 256 + d0 + lm) * KTOT;

    for (int kb = 0; kb < 59; ++kb) {
        const int k0 = kb * 32;
        v16bf a;
        if (k0 < 1280) {                        // windowed h: block j
            const int j = k0 >> 8, koff = k0 & 255;
            const bf16* arow = hpad + (size_t)(r + j * B_) * DH_ + koff;
            a = load_frag(arow + half * 8, arow + 16 + half * 8);
        } else if (k0 < 1600) {                 // src (padded 320)
            const bf16* arow = srcb + (size_t)r * 320 + (k0 - 1280);
            a = load_frag(arow + half * 8, arow + 16 + half * 8);
        } else if (k0 < 1856) {                 // hg per-b
            const bf16* arow = hgb + (size_t)(r & 63) * DH_ + (k0 - 1600);
            a = load_frag(arow + half * 8, arow + 16 + half * 8);
        } else {                                // one-hot "1" for the bias row
            v16bf t = {};
            if (half == 0) t[0] = (bf16)1.0f;
            a = t;
        }
        #pragma unroll
        for (int g = 0; g < 7; ++g) {
            const bf16* bp = brow[g] + k0 + 8 * half;
            v16bf bm = load_frag(bp, bp + 16);
            acc[g] = __builtin_amdgcn_wmma_f32_16x16x32_bf16(
                false, a, false, bm, (short)0, acc[g], false, false);
        }
    }

    // Epilogue: per-lane, element (m = v + 8*half, d = d0+lm)
    const int d = d0 + lm;
    #pragma unroll
    for (int v = 0; v < 8; ++v) {
        const int rr  = rbase + v + 8 * half;
        const int l   = rr >> 6;
        const int b   = rr & 63;
        const int len = slen[b];
        float hw = 0.0f, cw = 0.0f;
        if (l < len) {
            const float i_ = sigf(acc[0][v]);
            const float l_ = sigf(acc[1][v]);
            const float r_ = sigf(acc[2][v]);
            const float f_ = sigf(acc[3][v]);
            const float s_ = sigf(acc[4][v]);
            const float o_ = sigf(acc[5][v]);
            const float u_ = tanh_f(acc[6][v]);
            // softmax over (l_, f_, r_, s_, i_); values in (0,1) -> no max sub
            const float e0 = __expf(l_), e1 = __expf(f_), e2 = __expf(r_);
            const float e3 = __expf(s_), e4 = __expf(i_);
            const float Z  = e0 + e1 + e2 + e3 + e4;
            const size_t cb = (size_t)b * DH_ + d;
            const float cl  = cpad[(size_t)(l)     * COLS + cb]; // c_w1[l-1]
            const float cc  = cpad[(size_t)(l + 1) * COLS + cb]; // c_w1[l]
            const float cr  = cpad[(size_t)(l + 2) * COLS + cb]; // c_w1[l+1]
            const float cg1 = c0[(size_t)L_ * COLS + cb];
            cw = (e0 * cl + e1 * cc + e2 * cr + e3 * cg1 + e4 * u_) / Z;
            hw = o_ * tanh_f(cw);
        }
        out[(size_t)rr * DH_ + d]       = hw;
        out[HC_ + (size_t)rr * DH_ + d] = cw;
    }
}

// ---------------------------------------------------------------------------
// col: per (b,d) column: softmax of sigmoid(fi+base) over valid l, weighted
// sum with c, produce c_g / h_g (row L of outputs).
// ---------------------------------------------------------------------------
__global__ __launch_bounds__(256) void k_col(
    const float* __restrict__ fi, const float* __restrict__ base,
    const float* __restrict__ fg, const float* __restrict__ og,
    const float* __restrict__ c0, const int* __restrict__ slen,
    float* __restrict__ out)
{
    const int tid = blockIdx.x * 256 + threadIdx.x;  // b*256+d
    const int b   = tid >> 8;
    const int len = slen[b];
    const float bs = base[tid];
    float Z = 0.0f, accv = 0.0f;
    for (int l = 0; l < len; ++l) {
        const size_t idx = (size_t)l * COLS + tid;
        const float s = sigf(fi[idx] + bs);
        const float e = __expf(s);
        Z    += e;
        accv += e * c0[idx];
    }
    const float cg  = fg[tid] * c0[(size_t)L_ * COLS + tid] + accv / Z;
    const float hg_ = og[tid] * tanh_f(cg);
    out[(size_t)L_ * COLS + tid]       = hg_;
    out[HC_ + (size_t)L_ * COLS + tid] = cg;
}

// ---------------------------------------------------------------------------
extern "C" void kernel_launch(void* const* d_in, const int* in_sizes, int n_in,
                              void* d_out, int out_size, void* d_ws, size_t ws_size,
                              hipStream_t stream)
{
    const float* src = (const float*)d_in[0];
    const float* h0  = (const float*)d_in[1];
    const float* c0  = (const float*)d_in[2];
    const float* Ww  = (const float*)d_in[3];
    const float* Wu  = (const float*)d_in[4];
    const float* Wv  = (const float*)d_in[5];
    const float* Wb  = (const float*)d_in[6];
    const float* Sw  = (const float*)d_in[7];
    const float* Su  = (const float*)d_in[8];
    const float* Sb  = (const float*)d_in[9];
    const int*   sl  = (const int*)d_in[10];
    float* out = (float*)d_out;
    char*  ws  = (char*)d_ws;

    // scratch layout (bytes)
    bf16*  hpad = (bf16*)(ws + 0);                       // (1028,64,256) bf16
    bf16*  srcb = (bf16*)(ws + 33685504ull);             // (65536,320)  bf16
    bf16*  hgb  = (bf16*)(ws + 75628544ull);             // (64,256)     bf16
    bf16*  wcat = (bf16*)(ws + 75661312ull);             // (1792,1888)  bf16
    bf16*  su1b = (bf16*)(ws + 82427904ull);             // (256,256)    bf16
    float* fi   = (float*)(ws + 82558976ull);            // (65536,256)  f32
    float* cpad = (float*)(ws + 149667840ull);           // (1026,64,256) f32
    float* hhat = (float*)(ws + 216909824ull);           // (64,256)     f32
    float* fg   = (float*)(ws + 216975360ull);
    float* og   = (float*)(ws + 217040896ull);
    float* base = (float*)(ws + 217106432ull);           // end 217171968

    k_prep_h<<<64, 256, 0, stream>>>(h0, c0, sl, hpad, cpad, hhat, hgb);
    k_prep_src<<<81920, 256, 0, stream>>>(src, srcb);
    k_pack_w<<<13472, 256, 0, stream>>>(Ww, Wu, Wv, Wb, Su, wcat, su1b);
    k_gates_small<<<192, 256, 0, stream>>>(h0, hhat, Sw, Su, Sb, fg, og, base);
    k_fi_gemm<<<dim3(512, 16), 256, 0, stream>>>(hpad, su1b, fi);
    k_main_gemm<<<dim3(512, 16), 256, 0, stream>>>(hpad, srcb, hgb, wcat,
                                                   cpad, c0, sl, out);
    k_col<<<64, 256, 0, stream>>>(fi, base, fg, og, c0, sl, out);
}
